// CRNModel_53996328845725
// MI455X (gfx1250) — matrix operand order repaired and checked by reading, not compiled, not run on hardware
//
#include <hip/hip_runtime.h>
#include <hip/hip_bf16.h>
#include <math.h>

typedef __attribute__((ext_vector_type(16))) _Float16 v16h;
typedef __attribute__((ext_vector_type(8)))  float    v8f;

#define LIP_C 1.9679896712654303f   // 15^(1/4)

__device__ __forceinline__ v8f wmma_f16(v16h a, v16h b, v8f c) {
  // D = A(16x32 f16) * B(32x16 f16) + C(16x16 f32)
  return __builtin_amdgcn_wmma_f32_16x16x32_f16(false, a, false, b, (short)0, c, false, false);
}

// A-fragment (16x32, f16) from row-major LDS [16][ld] at K offset kofs.
// Lane l<16: row=l, K = kofs+{0..7,16..23}; lane l>=16: row=l-16, K = kofs+{8..15,24..31}.
__device__ __forceinline__ v16h ldsA(const _Float16* s, int ld, int kofs, int lane) {
  const _Float16* p = s + (lane & 15) * ld + kofs + ((lane < 16) ? 0 : 8);
  v16h a;
#pragma unroll
  for (int e = 0; e < 8; ++e) { a[e] = p[e]; a[e + 8] = p[e + 16]; }
  return a;
}

// B-fragment (32x16, f16) for B = W^T, W stored row-major [N][ldk] in LDS.
// Lane l: column n = nofs + (l&15); K = kofs + (l<16 ? {0..7,16..23} : {8..15,24..31}).
__device__ __forceinline__ v16h ldsB(const _Float16* w, int ldk, int nofs, int kofs, int lane) {
  const _Float16* p = w + (nofs + (lane & 15)) * ldk + kofs + ((lane < 16) ? 0 : 8);
  v16h b;
#pragma unroll
  for (int e = 0; e < 8; ++e) { b[e] = p[e]; b[e + 8] = p[e + 16]; }
  return b;
}

// fast transcendentals: v_rcp_f32 / v_exp_f32 / (v_tanh_f32 when exposed)
__device__ __forceinline__ float fast_rcp(float x) { return __builtin_amdgcn_rcpf(x); }
__device__ __forceinline__ float sigm(float x) {
  return fast_rcp(1.0f + __expf(-x));
}
__device__ __forceinline__ float fast_tanh(float x) {
#if __has_builtin(__builtin_amdgcn_tanhf)
  return __builtin_amdgcn_tanhf(x);
#else
  // tanh(x) = 2 / (1 + e^(-2x)) - 1
  return 2.0f * fast_rcp(1.0f + __expf(-2.0f * x)) - 1.0f;
#endif
}
__device__ __forceinline__ float elu1(float x) { return x > 0.0f ? x : (__expf(x) - 1.0f); }

// ---------- Pass 1: Lipschitz scales: s = min(1, c / (||W||_F + 1e-12)) ----------
__global__ void crn_lip_scales(const float* __restrict__ Wih, const float* __restrict__ Whh,
                               const float* __restrict__ Wbr, const float* __restrict__ Wo1,
                               const float* __restrict__ Wo2, float* __restrict__ scales) {
  __shared__ float red[256];
  const float* ws[5] = {Wih, Whh, Wbr, Wo1, Wo2};
  const int    ns[5] = {512 * 32, 512 * 128, 64 * 128, 128 * 68, 128};
  const int b = blockIdx.x;
  const float* w = ws[b];
  const int n = ns[b];
  float ss = 0.0f;
  for (int i = threadIdx.x; i < n; i += 256) { float x = w[i]; ss += x * x; }
  red[threadIdx.x] = ss;
  __syncthreads();
  for (int s = 128; s > 0; s >>= 1) {
    if (threadIdx.x < s) red[threadIdx.x] += red[threadIdx.x + s];
    __syncthreads();
  }
  if (threadIdx.x == 0) scales[b] = fminf(1.0f, LIP_C / (sqrtf(red[0]) + 1e-12f));
}

// ---------- Pass 2: full CRN forward. 64 WGs x 4 waves; each wave owns 16 batch rows ----------
__global__ __launch_bounds__(128, 1) void crn_fwd_kernel(
    const float* __restrict__ cov, const float* __restrict__ trt,
    const float* __restrict__ h0,  const float* __restrict__ c0,
    const float* __restrict__ Wih, const float* __restrict__ Whh,
    const float* __restrict__ bih, const float* __restrict__ bhh,
    const float* __restrict__ Wbr, const float* __restrict__ bbr,
    const float* __restrict__ Wt1, const float* __restrict__ bt1,
    const float* __restrict__ Wt2, const float* __restrict__ bt2,
    const float* __restrict__ Wo1, const float* __restrict__ bo1,
    const float* __restrict__ Wo2, const float* __restrict__ bo2,
    const float* __restrict__ scales, float* __restrict__ out)
{
  // weights (f16, pre-scaled, padded) — ~225 KB of the 320 KB/WGP LDS
  __shared__ _Float16 sWih[512 * 32];   // [4H][C]
  __shared__ _Float16 sWhh[512 * 128];  // [4H][H]
  __shared__ _Float16 sWbr[64 * 128];   // [BR][H]
  __shared__ _Float16 sWt1[128 * 64];   // [FC][BR]
  __shared__ _Float16 sWt2[16 * 128];   // [NT->16][FC], rows 4..15 zero
  __shared__ _Float16 sWo1[128 * 96];   // [FC][68->96], cols 68..95 zero
  __shared__ _Float16 sWo2[16 * 128];   // [NO->16][FC], rows 1..15 zero
  __shared__ float sBz[512];            // b_ih + b_hh
  __shared__ float sBbr[64], sBt1[128], sBt2[16], sBo1[128], sBo2[16];
  // per-wave staging
  __shared__ _Float16 sH[4][16 * 128];  // hidden state h_t (row-major f16)
  __shared__ _Float16 sT[4][16 * 128];  // activation staging (br / tl1 / o1)
  __shared__ _Float16 sX[4][16 * 32];   // treatment A-tile (cols 4..31 zero)
  __shared__ float    sTL[4][16 * 4];   // treatment logits for softmax
  __shared__ float    sPrev[4][16];     // outcome feedback

  const int tid = threadIdx.x;
  const int wave = tid >> 5, lane = tid & 31;
  const int hi = (lane >= 16) ? 8 : 0;      // C-layout row offset for upper half-wave
  const int nl = lane & 15;                 // C-layout column
  const int rowbase = blockIdx.x * 64 + wave * 16;

  const float s_ih = scales[0], s_hh = scales[1], s_br = scales[2];
  const float s_o1 = scales[3], s_o2 = scales[4];

  for (int i = tid; i < 512 * 32;  i += 128) sWih[i] = (_Float16)(Wih[i] * s_ih);
  for (int i = tid; i < 512 * 128; i += 128) sWhh[i] = (_Float16)(Whh[i] * s_hh);
  for (int i = tid; i < 64 * 128;  i += 128) sWbr[i] = (_Float16)(Wbr[i] * s_br);
  for (int i = tid; i < 128 * 64;  i += 128) sWt1[i] = (_Float16)Wt1[i];
  for (int i = tid; i < 16 * 128;  i += 128) sWt2[i] = (_Float16)((i >> 7) < 4 ? Wt2[i] : 0.0f);
  for (int i = tid; i < 128 * 96;  i += 128) {
    int r = i / 96, k = i % 96;
    sWo1[i] = (_Float16)(k < 68 ? Wo1[r * 68 + k] * s_o1 : 0.0f);
  }
  for (int i = tid; i < 16 * 128;  i += 128) sWo2[i] = (_Float16)(i < 128 ? Wo2[i] * s_o2 : 0.0f);
  for (int i = tid; i < 512; i += 128) sBz[i] = bih[i] + bhh[i];
  if (tid < 64)  sBbr[tid] = bbr[tid];
  if (tid < 128) sBt1[tid] = bt1[tid];
  if (tid < 16)  sBt2[tid] = (tid < 4) ? bt2[tid] : 0.0f;
  if (tid < 128) sBo1[tid] = bo1[tid];
  if (tid < 16)  sBo2[tid] = (tid == 0) ? bo2[0] : 0.0f;

  // h0 -> per-wave LDS stage (f16); c0 -> C-layout accumulator registers
  for (int i = lane; i < 16 * 128; i += 32) {
    int r = i >> 7, cc = i & 127;
    sH[wave][i] = (_Float16)h0[(size_t)(rowbase + r) * 128 + cc];
  }
  v8f cfr[8];
#pragma unroll
  for (int jt = 0; jt < 8; ++jt)
#pragma unroll
    for (int v = 0; v < 8; ++v)
      cfr[jt][v] = c0[(size_t)(rowbase + v + hi) * 128 + jt * 16 + nl];

  __syncthreads();

  float* out_bal = out;                 // [4096][64][64]
  float* out_tpr = out + 16777216;      // [4096][64][4]
  float* out_oc  = out + 17825792;      // [4096][64][1]
  float* out_h   = out + 18087936;      // [4096][128]
  float* out_c   = out + 18612224;      // [4096][128]

  for (int t = 0; t < 64; ++t) {
    // ---- x_t A-fragment; feedback of outcome into covariate column 27 (lane>=16, K=24..31 -> elem 11)
    v16h ax;
    {
      const float* cp = cov + ((size_t)(rowbase + nl) * 64 + t) * 32 + ((lane < 16) ? 0 : 8);
#pragma unroll
      for (int e = 0; e < 8; ++e) { ax[e] = (_Float16)cp[e]; ax[e + 8] = (_Float16)cp[e + 16]; }
      __builtin_prefetch(cp + 32, 0, 0);  // next timestep's covariates -> global_prefetch
      if (t > 0 && lane >= 16) ax[11] = (_Float16)sPrev[wave][nl];
    }
    // ---- h_{t-1} A-fragments
    v16h hfr[4];
#pragma unroll
    for (int kt = 0; kt < 4; ++kt) hfr[kt] = ldsA(sH[wave], 128, kt * 32, lane);

    // ---- LSTM: z = x@Wih^T + h@Whh^T + b, gate fusion per 16-column tile
#pragma unroll
    for (int jt = 0; jt < 8; ++jt) {
      v8f acc[4];
#pragma unroll
      for (int g = 0; g < 4; ++g) {
        const int n = g * 128 + jt * 16;
        const float bz = sBz[n + nl];
#pragma unroll
        for (int v = 0; v < 8; ++v) acc[g][v] = bz;
        acc[g] = wmma_f16(ax, ldsB(sWih, 32, n, 0, lane), acc[g]);
#pragma unroll
        for (int kt = 0; kt < 4; ++kt)
          acc[g] = wmma_f16(hfr[kt], ldsB(sWhh, 128, n, kt * 32, lane), acc[g]);
      }
      const int ccol = jt * 16 + nl;
#pragma unroll
      for (int v = 0; v < 8; ++v) {
        float ig = sigm(acc[0][v]), fg = sigm(acc[1][v]);
        float gg = fast_tanh(acc[2][v]), og = sigm(acc[3][v]);
        float cn = fg * cfr[jt][v] + ig * gg;
        cfr[jt][v] = cn;
        float hn = og * fast_tanh(cn);
        sH[wave][(v + hi) * 128 + ccol] = (_Float16)hn;
        if (t == 63) out_h[(size_t)(rowbase + v + hi) * 128 + ccol] = hn;
      }
    }
    asm volatile("s_wait_dscnt 0x0" ::: "memory");
#pragma unroll
    for (int kt = 0; kt < 4; ++kt) hfr[kt] = ldsA(sH[wave], 128, kt * 32, lane);  // h_t

    // ---- br = elu(h @ Wbr^T + b) : store to bal output + stage as f16
#pragma unroll
    for (int nt = 0; nt < 4; ++nt) {
      const int n = nt * 16;
      v8f a; const float bb = sBbr[n + nl];
#pragma unroll
      for (int v = 0; v < 8; ++v) a[v] = bb;
#pragma unroll
      for (int kt = 0; kt < 4; ++kt)
        a = wmma_f16(hfr[kt], ldsB(sWbr, 128, n, kt * 32, lane), a);
#pragma unroll
      for (int v = 0; v < 8; ++v) {
        float e = elu1(a[v]);
        out_bal[((size_t)(rowbase + v + hi) * 64 + t) * 64 + n + nl] = e;
        sT[wave][(v + hi) * 64 + n + nl] = (_Float16)e;
      }
    }
    asm volatile("s_wait_dscnt 0x0" ::: "memory");
    v16h brA0 = ldsA(sT[wave], 64, 0, lane);
    v16h brA1 = ldsA(sT[wave], 64, 32, lane);

    // ---- treatment head: tl1 = elu(br@Wt1^T+b); tl = tl1@Wt2^T+b; softmax over 4 cols
#pragma unroll
    for (int nt = 0; nt < 8; ++nt) {
      const int n = nt * 16;
      v8f a; const float bb = sBt1[n + nl];
#pragma unroll
      for (int v = 0; v < 8; ++v) a[v] = bb;
      a = wmma_f16(brA0, ldsB(sWt1, 64, n, 0, lane), a);
      a = wmma_f16(brA1, ldsB(sWt1, 64, n, 32, lane), a);
#pragma unroll
      for (int v = 0; v < 8; ++v)
        sT[wave][(v + hi) * 128 + n + nl] = (_Float16)elu1(a[v]);
    }
    asm volatile("s_wait_dscnt 0x0" ::: "memory");
    {
      v8f tl; const float bb = sBt2[nl];
#pragma unroll
      for (int v = 0; v < 8; ++v) tl[v] = bb;
#pragma unroll
      for (int kt = 0; kt < 4; ++kt) {
        v16h aa = ldsA(sT[wave], 128, kt * 32, lane);
        tl = wmma_f16(aa, ldsB(sWt2, 128, 0, kt * 32, lane), tl);
      }
      if (nl < 4) {
#pragma unroll
        for (int v = 0; v < 8; ++v) sTL[wave][(v + hi) * 4 + nl] = tl[v];
      }
    }
    asm volatile("s_wait_dscnt 0x0" ::: "memory");
    if (lane < 16) {
      const float* r = &sTL[wave][lane * 4];
      float m = fmaxf(fmaxf(r[0], r[1]), fmaxf(r[2], r[3]));
      float e0 = __expf(r[0] - m), e1 = __expf(r[1] - m);
      float e2 = __expf(r[2] - m), e3 = __expf(r[3] - m);
      float inv = fast_rcp(e0 + e1 + e2 + e3);
      float* tp = out_tpr + ((size_t)(rowbase + lane) * 64 + t) * 4;
      tp[0] = e0 * inv; tp[1] = e1 * inv; tp[2] = e2 * inv; tp[3] = e3 * inv;
    }

    // ---- outcome head: oc = elu([br,treat]@Wo1^T+b)@Wo2^T+b  (K padded 68->96)
    for (int i = lane; i < 16 * 32; i += 32) {
      int r = i >> 5, cc = i & 31;
      sX[wave][i] = (_Float16)(cc < 4 ? trt[((size_t)(rowbase + r) * 64 + t) * 4 + cc] : 0.0f);
    }
    asm volatile("s_wait_dscnt 0x0" ::: "memory");
    v16h trA = ldsA(sX[wave], 32, 0, lane);
#pragma unroll
    for (int nt = 0; nt < 8; ++nt) {
      const int n = nt * 16;
      v8f a; const float bb = sBo1[n + nl];
#pragma unroll
      for (int v = 0; v < 8; ++v) a[v] = bb;
      a = wmma_f16(brA0, ldsB(sWo1, 96, n, 0, lane), a);
      a = wmma_f16(brA1, ldsB(sWo1, 96, n, 32, lane), a);
      a = wmma_f16(trA,  ldsB(sWo1, 96, n, 64, lane), a);
#pragma unroll
      for (int v = 0; v < 8; ++v)
        sT[wave][(v + hi) * 128 + n + nl] = (_Float16)elu1(a[v]);
    }
    asm volatile("s_wait_dscnt 0x0" ::: "memory");
    {
      v8f oc; const float bb = sBo2[nl];
#pragma unroll
      for (int v = 0; v < 8; ++v) oc[v] = bb;
#pragma unroll
      for (int kt = 0; kt < 4; ++kt) {
        v16h aa = ldsA(sT[wave], 128, kt * 32, lane);
        oc = wmma_f16(aa, ldsB(sWo2, 128, 0, kt * 32, lane), oc);
      }
      if (nl == 0) {
#pragma unroll
        for (int v = 0; v < 8; ++v) {
          sPrev[wave][v + hi] = oc[v];
          out_oc[(size_t)(rowbase + v + hi) * 64 + t] = oc[v];
        }
      }
    }
    asm volatile("s_wait_dscnt 0x0" ::: "memory");
  }

  // final cell state
#pragma unroll
  for (int jt = 0; jt < 8; ++jt) {
    const int col = jt * 16 + nl;
#pragma unroll
    for (int v = 0; v < 8; ++v)
      out_c[(size_t)(rowbase + v + hi) * 128 + col] = cfr[jt][v];
  }
}

extern "C" void kernel_launch(void* const* d_in, const int* in_sizes, int n_in,
                              void* d_out, int out_size, void* d_ws, size_t ws_size,
                              hipStream_t stream) {
  const float* cov = (const float*)d_in[0];
  const float* trt = (const float*)d_in[1];
  const float* h0  = (const float*)d_in[2];
  const float* c0  = (const float*)d_in[3];
  // d_in[4] = alpha: only affects the backward pass (flip_grad is identity forward)
  const float* Wih = (const float*)d_in[5];
  const float* Whh = (const float*)d_in[6];
  const float* bih = (const float*)d_in[7];
  const float* bhh = (const float*)d_in[8];
  const float* Wbr = (const float*)d_in[9];
  const float* bbr = (const float*)d_in[10];
  const float* Wt1 = (const float*)d_in[11];
  const float* bt1 = (const float*)d_in[12];
  const float* Wt2 = (const float*)d_in[13];
  const float* bt2 = (const float*)d_in[14];
  const float* Wo1 = (const float*)d_in[15];
  const float* bo1 = (const float*)d_in[16];
  const float* Wo2 = (const float*)d_in[17];
  const float* bo2 = (const float*)d_in[18];
  float* scales = (float*)d_ws;

  crn_lip_scales<<<5, 256, 0, stream>>>(Wih, Whh, Wbr, Wo1, Wo2, scales);
  crn_fwd_kernel<<<64, 128, 0, stream>>>(cov, trt, h0, c0,
                                         Wih, Whh, bih, bhh, Wbr, bbr,
                                         Wt1, bt1, Wt2, bt2, Wo1, bo1, Wo2, bo2,
                                         scales, (float*)d_out);
}
